// PINN_g_rr_63917703299594
// MI455X (gfx1250) — matrix operand order repaired and checked by reading, not compiled, not run on hardware
//
#include <hip/hip_runtime.h>
#include <hip/hip_bf16.h>
#include <math.h>

// ---------------------------------------------------------------------------
// PINN Einstein-equation loss, diagonal Schwarzschild-like ansatz.
// Heavy part: value + 4 first-order + 10 second-order forward-mode AD rows
// through the 256x256 hidden layer -> per point a 16x256 @ 256x256 bf16 GEMM
// on v_wmma_f32_16x16x32_bf16 (f32 accumulate), A and B both fed from LDS,
// B double-buffered in registers to overlap ds_load latency with the XDL pipe.
// ---------------------------------------------------------------------------

typedef __attribute__((ext_vector_type(16))) __bf16 v16bf;
typedef __attribute__((ext_vector_type(8)))  __bf16 bf16x8;
typedef __attribute__((ext_vector_type(8)))  float  v8f;

#define RSCH 0.02f
#define HDIM 256

union AV { v16bf v; bf16x8 h[2]; };

// ---- kernel 0: W2 (f32, [k][n]) -> W2T (bf16, [n][k]) ---------------------
__global__ void k_cvt_w2t(const float* __restrict__ W2, __bf16* __restrict__ W2T) {
  int idx = blockIdx.x * blockDim.x + threadIdx.x;   // 0..65535
  int k = idx >> 8, n = idx & 255;
  W2T[n * HDIM + k] = (__bf16)W2[k * HDIM + n];
}

// ---- kernel 1: zero the scalar output (harness does not re-poison) --------
__global__ void k_zero(float* out) {
  if (blockIdx.x == 0 && threadIdx.x == 0) out[0] = 0.0f;
}

// ---- kernel 2: per-point u, du[4], d2u[10] via WMMA ------------------------
// 4 waves/block, one point per wave per iteration.
// A-matrix rows: 0=h1, 1..4=dh1_a, 5..14=d2h1_ab, 15=pad (M=16 exactly).
__global__ void __launch_bounds__(128)
k_mlp_d2(const float* __restrict__ coords,
         const float* __restrict__ W1, const float* __restrict__ b1,
         const __bf16* __restrict__ W2T, const float* __restrict__ b2,
         const float* __restrict__ W3, const float* __restrict__ b3,
         float* __restrict__ rec, int npts)
{
  __shared__ __align__(16) __bf16 sW2T[HDIM * HDIM];   // 128 KB, block-shared
  __shared__ __align__(16) __bf16 sA[4][16 * HDIM];    // 8 KB per wave

  const int lane = threadIdx.x & 31;
  const int wv   = threadIdx.x >> 5;
  __bf16* A = sA[wv];
  const int  lm   = lane & 15;        // tile row (A) / column (B/D) index
  const bool hihf = lane >= 16;       // which K-half this lane holds
  const int  aoff = hihf ? 16 : 0;    // A: K offset
  const int  boff = hihf ? 16 : 0;    // B: K offset

  // ---- stage W2T into LDS once per block --------------------------------
  {
    const bf16x8* src = (const bf16x8*)W2T;
    bf16x8*       dst = (bf16x8*)sW2T;
    for (int i = threadIdx.x; i < (HDIM * HDIM) / 8; i += blockDim.x)
      dst[i] = src[i];
  }
  __syncthreads();

  // ---- hoist point-invariant layer-1 weights into registers ---------------
  float w1r[8][4], b1r[8];
  #pragma unroll
  for (int i = 0; i < 8; ++i) {
    const int hn = lane * 8 + i;
    b1r[i] = b1[hn];
    #pragma unroll
    for (int a = 0; a < 4; ++a) w1r[i][a] = W1[a * HDIM + hn];
  }

  const int gw = blockIdx.x * 4 + wv;
  const int nw = gridDim.x * 4;

  for (int p = gw; p < npts; p += nw) {
    const float4 c = *(const float4*)(coords + 4 * p);

    // ----- layer 1 + exact 1st/2nd derivatives (z1 is linear in c) --------
    #pragma unroll
    for (int i = 0; i < 8; ++i) {
      const int hn = lane * 8 + i;
      const float z = b1r[i] + c.x * w1r[i][0] + c.y * w1r[i][1]
                             + c.z * w1r[i][2] + c.w * w1r[i][3];
      const float h1 = tanhf(z);
      const float s  = 1.0f - h1 * h1;
      A[0 * HDIM + hn] = (__bf16)h1;
      #pragma unroll
      for (int a = 0; a < 4; ++a) A[(1 + a) * HDIM + hn] = (__bf16)(s * w1r[i][a]);
      const float c2 = -2.0f * h1 * s * s;            // d2h1_ab = c2*wa*wb
      int ridx = 5;
      #pragma unroll
      for (int a = 0; a < 4; ++a)
        #pragma unroll
        for (int b = a; b < 4; ++b) {
          A[ridx * HDIM + hn] = (__bf16)(c2 * w1r[i][a] * w1r[i][b]);
          ++ridx;
        }
      A[15 * HDIM + hn] = (__bf16)0.0f;               // pad row -> M=16
    }
    // per-wave produce->consume through LDS: DS pipe is in-order per wave.

    // ----- Z2 tile-by-tile + fused layer-2/3, no Z round-trip --------------
    float vals[15];
    #pragma unroll
    for (int t = 0; t < 15; ++t) vals[t] = 0.0f;

    #pragma unroll 1
    for (int nt = 0; nt < 16; ++nt) {
      const int j = nt * 16 + lm;
      const __bf16* bcol = sW2T + (nt * 16 + lm) * HDIM + boff;
      // scalar operands for the tail: issue early, consumed after the burst
      const float b2j = b2[j];
      const float w3v = W3[j];

      v8f acc0 = {}, acc1 = {};           // 2 chains: hide XDL latency
      // software-pipelined B: distinct cur/next registers so ds_loads of the
      // next step overlap with the current WMMAs (no s_wait_dscnt 0 chain)
      AV b0c, b1c;
      b0c.h[0] = *(const bf16x8*)(bcol + 0);
      b0c.h[1] = *(const bf16x8*)(bcol + 8);
      b1c.h[0] = *(const bf16x8*)(bcol + 32);
      b1c.h[1] = *(const bf16x8*)(bcol + 40);
      #pragma unroll
      for (int kk = 0; kk < 8; kk += 2) {
        const int k0 = kk * 32;
        AV b0n = b0c, b1n = b1c;
        if (kk < 6) {
          b0n.h[0] = *(const bf16x8*)(bcol + k0 + 64);
          b0n.h[1] = *(const bf16x8*)(bcol + k0 + 72);
          b1n.h[0] = *(const bf16x8*)(bcol + k0 + 96);
          b1n.h[1] = *(const bf16x8*)(bcol + k0 + 104);
        }
        AV a0, a1;                        // A re-read from LDS for every nt
        a0.h[0] = *(const bf16x8*)&A[lm * HDIM + k0 + aoff];
        a0.h[1] = *(const bf16x8*)&A[lm * HDIM + k0 + 8 + aoff];
        a1.h[0] = *(const bf16x8*)&A[lm * HDIM + k0 + 32 + aoff];
        a1.h[1] = *(const bf16x8*)&A[lm * HDIM + k0 + 40 + aoff];
        acc0 = __builtin_amdgcn_wmma_f32_16x16x32_bf16(
            false, a0.v, false, b0c.v, (short)0, acc0, false, false);
        acc1 = __builtin_amdgcn_wmma_f32_16x16x32_bf16(
            false, a1.v, false, b1c.v, (short)0, acc1, false, false);
        b0c = b0n; b1c = b1n;
      }
      // redistribute: lane L and L+16 hold complementary row halves of col N.
      // Static indices + value selects (no dynamic register-array indexing).
      float z[16];
      #pragma unroll
      for (int v = 0; v < 8; ++v) {
        const float s = acc0[v] + acc1[v];
        const float o = __shfl_xor(s, 16, 32);
        z[v]     = hihf ? o : s;   // rows 0..7 of this column
        z[v + 8] = hihf ? s : o;   // rows 8..15 of this column
      }
      // both half-waves now hold the same column -> reduce within 16 lanes
      const float z0 = z[0] + b2j;
      const float h2 = tanhf(z0);
      const float s2 = 1.0f - h2 * h2;
      vals[0] += h2 * w3v;
      float zp[4];
      #pragma unroll
      for (int a = 0; a < 4; ++a) {
        zp[a] = z[1 + a];
        vals[1 + a] += s2 * zp[a] * w3v;
      }
      const float c2 = -2.0f * h2 * s2 * s2;
      int ridx = 0;
      #pragma unroll
      for (int a = 0; a < 4; ++a)
        #pragma unroll
        for (int b = a; b < 4; ++b) {
          vals[5 + ridx] += (s2 * z[5 + ridx] + c2 * zp[a] * zp[b]) * w3v;
          ++ridx;
        }
    }

    // tree reduction over the 16 distinct columns (halves are duplicates)
    #pragma unroll
    for (int t = 0; t < 15; ++t) {
      #pragma unroll
      for (int m = 8; m >= 1; m >>= 1)
        vals[t] += __shfl_xor(vals[t], m, 32);
    }
    if (lane == 0) {
      rec[p * 16 + 0] = vals[0] + b3[0];
      #pragma unroll
      for (int t = 1; t < 15; ++t) rec[p * 16 + t] = vals[t];
      rec[p * 16 + 15] = 0.0f;
    }
  }
}

// ---- kernel 3: diagonal-metric Einstein tensor + mean-square loss ----------
__global__ void __launch_bounds__(256)
k_einstein(const float* __restrict__ coords, const float* __restrict__ rec,
           float* __restrict__ out, int npts, float scale)
{
  const int p = blockIdx.x * blockDim.x + threadIdx.x;
  float local = 0.0f;
  if (p < npts) {
    const float r  = coords[4 * p + 1];
    const float th = coords[4 * p + 2];
    const float u  = rec[16 * p + 0];
    float du[4];
    #pragma unroll
    for (int a = 0; a < 4; ++a) du[a] = rec[16 * p + 1 + a];
    const int base[4] = { 0, 3, 5, 6 };       // symmetric (a<=b) packing
    float d2u[4][4];
    #pragma unroll
    for (int a = 0; a < 4; ++a)
      #pragma unroll
      for (int b = 0; b < 4; ++b) {
        const int aa = a < b ? a : b, bb = a < b ? b : a;
        d2u[a][b] = rec[16 * p + 5 + base[aa] + bb];
      }

    // value / gradient / Hessian of each diagonal metric component
    float g[4], Dg[4][4] = {}, Hg[4][4][4] = {};
    const float eu = expf(u);
    const float sn = sinf(th), cs = cosf(th);
    g[0] = -(1.0f - RSCH / r);
    Dg[0][1] = -RSCH / (r * r);
    Hg[0][1][1] = 2.0f * RSCH / (r * r * r);
    g[1] = eu;
    #pragma unroll
    for (int a = 0; a < 4; ++a) {
      Dg[1][a] = eu * du[a];
      #pragma unroll
      for (int b = 0; b < 4; ++b) Hg[1][a][b] = eu * (du[a] * du[b] + d2u[a][b]);
    }
    g[2] = r * r;  Dg[2][1] = 2.0f * r;  Hg[2][1][1] = 2.0f;
    g[3] = r * r * sn * sn;
    Dg[3][1] = 2.0f * r * sn * sn;
    Dg[3][2] = 2.0f * r * r * sn * cs;
    Hg[3][1][1] = 2.0f * sn * sn;
    Hg[3][1][2] = Hg[3][2][1] = 4.0f * r * sn * cs;
    Hg[3][2][2] = 2.0f * r * r * (cs * cs - sn * sn);

    float gi[4];
    #pragma unroll
    for (int i = 0; i < 4; ++i) gi[i] = 1.0f / g[i];

    // Gamma^i_{jk} for diagonal metric
    float gam[4][4][4];
    #pragma unroll
    for (int i = 0; i < 4; ++i)
      #pragma unroll
      for (int j = 0; j < 4; ++j)
        #pragma unroll
        for (int k = 0; k < 4; ++k) {
          float t = 0.0f;
          if (k == i) t += Dg[i][j];
          if (j == i) t += Dg[i][k];
          if (j == k) t -= Dg[j][i];
          gam[i][j][k] = 0.5f * gi[i] * t;
        }

    auto dgam = [&](int i, int j, int k, int l) -> float {
      float t = 0.0f, ht = 0.0f;
      if (k == i) { t += Dg[i][j]; ht += Hg[i][j][l]; }
      if (j == i) { t += Dg[i][k]; ht += Hg[i][k][l]; }
      if (j == k) { t -= Dg[j][i]; ht -= Hg[j][i][l]; }
      return 0.5f * (-gi[i] * gi[i] * Dg[i][l] * t + gi[i] * ht);
    };

    // Ricci_{jk} = d_i G^i_jk - d_k G^i_ji + G^m_jk G^i_mi - G^m_ji G^i_mk
    float ric[4][4];
    #pragma unroll
    for (int j = 0; j < 4; ++j)
      #pragma unroll
      for (int k = 0; k < 4; ++k) {
        float s = 0.0f;
        #pragma unroll
        for (int i = 0; i < 4; ++i)
          s += dgam(i, j, k, i) - dgam(i, j, i, k);
        #pragma unroll
        for (int m = 0; m < 4; ++m)
          #pragma unroll
          for (int i = 0; i < 4; ++i)
            s += gam[m][j][k] * gam[i][m][i] - gam[m][j][i] * gam[i][m][k];
        ric[j][k] = s;
      }

    float Rsc = 0.0f;
    #pragma unroll
    for (int j = 0; j < 4; ++j) Rsc += gi[j] * ric[j][j];

    #pragma unroll
    for (int i = 0; i < 4; ++i)
      #pragma unroll
      for (int j = 0; j < 4; ++j) {
        float G = gi[i] * gi[j] * ric[i][j];
        if (i == j) G -= 0.5f * gi[i] * Rsc;
        local += G * G;
      }
  }

  __shared__ float red[256];
  red[threadIdx.x] = local;
  __syncthreads();
  #pragma unroll
  for (int s = 128; s >= 1; s >>= 1) {
    if (threadIdx.x < s) red[threadIdx.x] += red[threadIdx.x + s];
    __syncthreads();
  }
  if (threadIdx.x == 0) atomicAdd(out, red[0] * scale);
}

// ---------------------------------------------------------------------------
extern "C" void kernel_launch(void* const* d_in, const int* in_sizes, int n_in,
                              void* d_out, int out_size, void* d_ws, size_t ws_size,
                              hipStream_t stream)
{
  const float* coords = (const float*)d_in[0];
  const float* W1 = (const float*)d_in[1];
  const float* b1 = (const float*)d_in[2];
  const float* W2 = (const float*)d_in[3];
  const float* b2 = (const float*)d_in[4];
  const float* W3 = (const float*)d_in[5];
  const float* b3 = (const float*)d_in[6];
  const int npts = in_sizes[0] / 4;

  __bf16* W2T = (__bf16*)d_ws;                                        // 128 KB
  float*  rec = (float*)((char*)d_ws + HDIM * HDIM * sizeof(__bf16)); // npts*16 f32

  k_cvt_w2t<<<HDIM, HDIM, 0, stream>>>(W2, W2T);
  k_zero<<<1, 64, 0, stream>>>((float*)d_out);
  k_mlp_d2<<<2048, 128, 0, stream>>>(coords, W1, b1, W2T, b2, W3, b3, rec, npts);
  k_einstein<<<(npts + 255) / 256, 256, 0, stream>>>(
      coords, rec, (float*)d_out, npts, 1.0f / (npts * 16.0f));
}
